// MambaModularAddition_80255758893798
// MI455X (gfx1250) — matrix operand order repaired and checked by reading, not compiled, not run on hardware
//
#include <hip/hip_runtime.h>
#include <cstdint>

#define V_ 128
#define D_ 512
#define L_ 4
#define B_ 64
#define S_ 1024
#define H_ 1024
#define G_ 4096
#define M_ (B_*S_)
#define RMS_EPS 1.1920929e-07f
#define LN_EPS 1e-05f

typedef __attribute__((ext_vector_type(16))) __bf16 v16bf;
typedef __attribute__((ext_vector_type(8)))  float  v8f;
typedef __attribute__((ext_vector_type(4)))  unsigned int u32x4;
typedef __attribute__((ext_vector_type(8)))  int i32x8;
typedef __attribute__((ext_vector_type(4)))  int i32x4;

union ABFrag { uint4 q[2]; v16bf v; };

__device__ __forceinline__ unsigned short f2bf(float f) {
  unsigned u = __float_as_uint(f);
  u += 0x7FFFu + ((u >> 16) & 1u);          // round-to-nearest-even
  return (unsigned short)(u >> 16);
}

__device__ __forceinline__ float sigf(float x) { return 1.0f / (1.0f + __expf(-x)); }

// Opaque "writer" for TDM-staged LDS: the tensor_load_to_lds builtin carries no
// pointer operand, so without this the optimizer proves the LDS arrays are never
// stored to and folds all ds loads away (observed in round 2). Passing the LDS
// pointer through volatile asm with a memory clobber makes it escape.
__device__ __forceinline__ void lds_publish(void* p) {
  asm volatile("" : "+v"(p) : : "memory");
}

// ---------------- Tensor Data Mover staging ----------------
// D# per CDNA5 ISA ch.8: group0 = {count, lds_addr, global_addr, type=2},
// group1 = {data_size=2B, tensor_dim0/1, tile_dim0/1, tensor_dim0_stride}.
// 2D tile: rows x cols bf16, contiguous rows (stride == cols).
#if __has_builtin(__builtin_amdgcn_tensor_load_to_lds)
#define HAS_TDM 1
__device__ __forceinline__ void tdm_load_rows_to_lds(unsigned lds_byte_off,
                                                     const void* gsrc,
                                                     unsigned rows, unsigned cols) {
  unsigned long long ga = (unsigned long long)(uintptr_t)gsrc;
  u32x4 g0;
  g0[0] = 1u;                                               // count=1, user mode
  g0[1] = lds_byte_off;                                     // lds_addr
  g0[2] = (unsigned)ga;                                     // global_addr[31:0]
  g0[3] = (unsigned)((ga >> 32) & 0x01FFFFFFu) | 0x80000000u; // addr[56:32] | type=2
  i32x8 g1;
  g1[0] = 0x00010000;                                       // wg_mask=0, data_size=2B
  g1[1] = (int)((cols & 0xFFFFu) << 16);                    // tensor_dim0[15:0]
  g1[2] = (int)(((cols >> 16) & 0xFFFFu) | ((rows & 0xFFFFu) << 16)); // tdim0 hi | tdim1 lo
  g1[3] = (int)(((rows >> 16) & 0xFFFFu) | ((cols & 0xFFFFu) << 16)); // tdim1 hi | tile_dim0
  g1[4] = (int)(rows & 0xFFFFu);                            // tile_dim1 (tile_dim2=0)
  g1[5] = (int)cols;                                        // tensor_dim0_stride lo
  g1[6] = 0;
  g1[7] = 0;
  i32x4 z4 = {};
#if __clang_major__ >= 23
  i32x8 z8 = {};
  __builtin_amdgcn_tensor_load_to_lds(g0, g1, z4, z4, z8, 0);
#else
  __builtin_amdgcn_tensor_load_to_lds(g0, g1, z4, z4, 0);
#endif
}
#else
#define HAS_TDM 0
#endif

__device__ __forceinline__ void wait_tdm() {
#if __has_builtin(__builtin_amdgcn_s_wait_tensorcnt)
  __builtin_amdgcn_s_wait_tensorcnt(0);
#endif
}

// A fragment: 16x32 bf16 (MxK). lane<16: row=lane, K {k0..k0+7, k0+16..k0+23}
//             lane>=16: row=lane-16, K {k0+8..k0+15, k0+24..k0+31}
__device__ __forceinline__ v16bf load_a_frag(const unsigned short* A, int lda,
                                             int m0, int k0, int lane) {
  const unsigned short* p =
      A + (size_t)(m0 + (lane & 15)) * lda + k0 + ((lane & 16) ? 8 : 0);
  ABFrag u;
  u.q[0] = *reinterpret_cast<const uint4*>(p);
  u.q[1] = *reinterpret_cast<const uint4*>(p + 16);
  return u.v;
}

// B fragment from row-major weights W[N][K]: lane&15 selects column n,
// lane<16: K {k0..k0+7, k0+16..k0+23}; lane>=16: K {k0+8..k0+15, k0+24..k0+31}
__device__ __forceinline__ v16bf load_b_frag(const unsigned short* Bw, int ldb,
                                             int n0, int k0, int lane) {
  const unsigned short* p =
      Bw + (size_t)(n0 + (lane & 15)) * ldb + k0 + ((lane & 16) ? 8 : 0);
  ABFrag u;
  u.q[0] = *reinterpret_cast<const uint4*>(p);
  u.q[1] = *reinterpret_cast<const uint4*>(p + 16);
  return u.v;
}

#define WMMA_BF16(a, b, c) \
  __builtin_amdgcn_wmma_f32_16x16x32_bf16(false, (a), false, (b), (short)0, (c), false, false)

// ---------------- elementwise / prep kernels ----------------

__global__ void k_embed(const int* __restrict__ tok, const float* __restrict__ emb,
                        float* __restrict__ x) {
  size_t i = (size_t)blockIdx.x * 256 + threadIdx.x;
  int row = (int)(i >> 9);          // / D_
  int d   = (int)(i & (D_ - 1));
  x[i] = emb[(size_t)tok[row] * D_ + d];
}

__global__ void k_f2bf(const float* __restrict__ src, unsigned short* __restrict__ dst, int n) {
  int i = blockIdx.x * 256 + threadIdx.x;
  if (i < n) dst[i] = f2bf(src[i]);
}

__global__ void k_biascomb(const float* __restrict__ bi, const float* __restrict__ bh,
                           float* __restrict__ out) {
  int i = blockIdx.x * 256 + threadIdx.x;
  out[i] = bi[i] + bh[i];
}

__global__ void k_rmsnorm(const float* __restrict__ x, const float* __restrict__ w,
                          float* __restrict__ nx, unsigned short* __restrict__ nxb) {
  __shared__ float red[256];
  size_t base = (size_t)blockIdx.x * D_;
  int t = threadIdx.x;
  float v0 = x[base + t], v1 = x[base + t + 256];
  red[t] = v0 * v0 + v1 * v1;
  __syncthreads();
  for (int sft = 128; sft > 0; sft >>= 1) {
    if (t < sft) red[t] += red[t + sft];
    __syncthreads();
  }
  float sc = rsqrtf(red[0] * (1.0f / D_) + RMS_EPS);
  float o0 = v0 * sc * w[t];
  float o1 = v1 * sc * w[t + 256];
  nx[base + t] = o0;        nx[base + t + 256] = o1;
  nxb[base + t] = f2bf(o0); nxb[base + t + 256] = f2bf(o1);
}

__global__ void k_add_layernorm(const float* __restrict__ outp, const float* __restrict__ nx,
                                const float* __restrict__ w, const float* __restrict__ b,
                                float* __restrict__ x) {
  __shared__ float rs[256], rq[256];
  size_t base = (size_t)blockIdx.x * D_;
  int t = threadIdx.x;
  float t0 = outp[base + t] + nx[base + t];
  float t1 = outp[base + t + 256] + nx[base + t + 256];
  rs[t] = t0 + t1;
  rq[t] = t0 * t0 + t1 * t1;
  __syncthreads();
  for (int sft = 128; sft > 0; sft >>= 1) {
    if (t < sft) { rs[t] += rs[t + sft]; rq[t] += rq[t + sft]; }
    __syncthreads();
  }
  float mu  = rs[0] * (1.0f / D_);
  float var = rq[0] * (1.0f / D_) - mu * mu;
  float sc  = rsqrtf(var + LN_EPS);
  x[base + t]       += (t0 - mu) * sc * w[t]       + b[t];
  x[base + t + 256] += (t1 - mu) * sc * w[t + 256] + b[t + 256];
}

__global__ void k_finalnorm(const float* __restrict__ x, const float* __restrict__ w,
                            unsigned short* __restrict__ lastb) {
  __shared__ float red[256];
  size_t base = ((size_t)blockIdx.x * S_ + (S_ - 1)) * D_;
  int t = threadIdx.x;
  float v0 = x[base + t], v1 = x[base + t + 256];
  red[t] = v0 * v0 + v1 * v1;
  __syncthreads();
  for (int sft = 128; sft > 0; sft >>= 1) {
    if (t < sft) red[t] += red[t + sft];
    __syncthreads();
  }
  float sc = rsqrtf(red[0] * (1.0f / D_) + RMS_EPS);
  lastb[(size_t)blockIdx.x * D_ + t]       = f2bf(v0 * sc * w[t]);
  lastb[(size_t)blockIdx.x * D_ + t + 256] = f2bf(v1 * sc * w[t + 256]);
}

// ---------------- WMMA GEMM: C(MxN,f32) = A(MxK,bf16) * W(NxK,bf16)^T + bias ----
// A rows are contiguous at every call site (lda == K). The block's 16xK A tile is
// staged once into LDS (TDM when available), then all 8 waves read it via ds loads.

__global__ __launch_bounds__(256)
void k_gemm_bf16(const unsigned short* __restrict__ A, int lda,
                 const unsigned short* __restrict__ Bw, int ldb,
                 const float* __restrict__ bias,
                 float* __restrict__ C, int ldc, int K) {
  __shared__ unsigned short Atile[16 * 1024];   // declared first -> LDS offset 0
  int lane = threadIdx.x & 31;
  int wave = threadIdx.x >> 5;
  int m0 = blockIdx.x * 16;
  int n0 = (blockIdx.y * 8 + wave) * 16;
#if HAS_TDM
  if (wave == 0) {
    tdm_load_rows_to_lds(0u, A + (size_t)m0 * lda, 16u, (unsigned)K);
    wait_tdm();
  }
#else
  for (int i = threadIdx.x; i < (16 * K) / 8; i += 256)
    ((uint4*)Atile)[i] = ((const uint4*)(A + (size_t)m0 * lda))[i];
#endif
  lds_publish(Atile);              // opaque writer: keep ds loads of Atile alive
  __syncthreads();
  v8f acc = {};
#pragma unroll 2
  for (int k0 = 0; k0 < K; k0 += 32) {
    v16bf a = load_a_frag(Atile, K, 0, k0, lane);
    v16bf b = load_b_frag(Bw, ldb, n0, k0, lane);
    acc = WMMA_BF16(a, b, acc);
  }
  int col = n0 + (lane & 15);
  float bv = bias[col];
  int rbase = m0 + ((lane >> 4) << 3);
#pragma unroll
  for (int r = 0; r < 8; ++r)
    C[(size_t)(rbase + r) * ldc + col] = acc[r] + bv;
}

// ---------------- persistent LSTM recurrence ----------------
// 64 workgroups; WG w owns hidden units j in [16w,16w+16). Its 64 W_hh rows
// (4 gate-quadrant blocks of 16 contiguous rows) are staged ONCE into LDS via
// TDM and reused for all 1024 steps. Per step: gates = h @ W_hh_slab^T (WMMA,
// A from global bf16 h, B from LDS), cell update with c in registers, h -> global,
// grid-wide sense barrier.

__global__ __launch_bounds__(256)
void k_lstm(const float* __restrict__ xg, const unsigned short* __restrict__ Whh,
            unsigned short* __restrict__ hcur, unsigned short* __restrict__ hall,
            unsigned* barcnt, unsigned* bargen) {
  __shared__ unsigned short WhhLds[4 * 16 * H_];   // 128 KB, declared first -> LDS offset 0
  __shared__ float gates[4][64][16];               // 16 KB
  int lane = threadIdx.x & 31;
  int wave = threadIdx.x >> 5;
  int jbase = blockIdx.x * 16;
  int m0  = (wave >> 1) * 16;      // batch tile (4 tiles cover B=64)
  int gqA = (wave & 1) * 2;        // gate quadrants (i,f) or (g,o)
  int gqB = gqA + 1;
  int jl = lane & 15;
  int rb = (lane >> 4) << 3;
  float c[4] = {0.f, 0.f, 0.f, 0.f};

#if HAS_TDM
  if (wave < 4) {                  // one TDM per gate quadrant, issued by waves 0..3
    int q = wave;
    tdm_load_rows_to_lds((unsigned)(q * 16 * H_ * 2),
                         Whh + (size_t)(q * H_ + jbase) * H_, 16u, (unsigned)H_);
    wait_tdm();
  }
#else
  for (int i = threadIdx.x; i < (4 * 16 * H_) / 8; i += 256) {
    int q = i >> 11;               // / (16*H_/8)
    int rem = i & 2047;
    ((uint4*)WhhLds)[i] =
        *(const uint4*)(Whh + (size_t)(q * H_ + jbase) * H_ + (size_t)rem * 8);
  }
#endif
  lds_publish(WhhLds);             // opaque writer: keep ds loads of WhhLds alive
  __syncthreads();

  for (int s = 0; s < S_; ++s) {
    v8f acc0 = {}, acc1 = {};
    for (int k0 = 0; k0 < H_; k0 += 32) {
      v16bf a  = load_a_frag(hcur, H_, m0, k0, lane);
      v16bf b0 = load_b_frag(WhhLds, H_, gqA * 16, k0, lane);   // LDS-resident weights
      v16bf b1 = load_b_frag(WhhLds, H_, gqB * 16, k0, lane);
      acc0 = WMMA_BF16(a, b0, acc0);
      acc1 = WMMA_BF16(a, b1, acc1);
    }
#pragma unroll
    for (int r = 0; r < 8; ++r) {
      gates[gqA][m0 + rb + r][jl] = acc0[r];
      gates[gqB][m0 + rb + r][jl] = acc1[r];
    }
    __syncthreads();

#pragma unroll
    for (int p = 0; p < 4; ++p) {
      int idx = threadIdx.x + p * 256;
      int m = idx >> 4;            // batch row 0..63
      int j = idx & 15;            // local hidden unit
      int jg = jbase + j;
      size_t xrow = ((size_t)m * S_ + s) * G_;
      float iv = gates[0][m][j] + xg[xrow + 0 * H_ + jg];
      float fv = gates[1][m][j] + xg[xrow + 1 * H_ + jg];
      float gv = gates[2][m][j] + xg[xrow + 2 * H_ + jg];
      float ov = gates[3][m][j] + xg[xrow + 3 * H_ + jg];
      float cc = sigf(fv) * c[p] + sigf(iv) * tanhf(gv);
      c[p] = cc;
      unsigned short hb = f2bf(sigf(ov) * tanhf(cc));
      hcur[(size_t)m * H_ + jg] = hb;
      hall[((size_t)m * S_ + s) * H_ + jg] = hb;
    }

    // grid-wide sense barrier (h must be visible to all WGs before next step)
    __threadfence();
    __syncthreads();
    if (threadIdx.x == 0) {
      unsigned g = __atomic_load_n(bargen, __ATOMIC_RELAXED);
      if (atomicAdd(barcnt, 1u) == gridDim.x - 1u) {
        __atomic_store_n(barcnt, 0u, __ATOMIC_RELAXED);
        __atomic_fetch_add(bargen, 1u, __ATOMIC_RELEASE);
      } else {
        while (__atomic_load_n(bargen, __ATOMIC_ACQUIRE) == g)
          __builtin_amdgcn_s_sleep(2);
      }
    }
    __syncthreads();
  }
}

// ---------------- host launch ----------------

extern "C" void kernel_launch(void* const* d_in, const int* in_sizes, int n_in,
                              void* d_out, int out_size, void* d_ws, size_t ws_size,
                              hipStream_t stream) {
  (void)in_sizes; (void)n_in; (void)out_size; (void)ws_size;
  const int*   tokens       = (const int*)d_in[0];
  const float* embedding    = (const float*)d_in[1];
  const float* pre_norm_w   = (const float*)d_in[2];
  const float* W_ih         = (const float*)d_in[3];
  const float* W_hh         = (const float*)d_in[4];
  const float* b_ih         = (const float*)d_in[5];
  const float* b_hh         = (const float*)d_in[6];
  const float* W_out        = (const float*)d_in[7];
  const float* b_out        = (const float*)d_in[8];
  const float* ln_w         = (const float*)d_in[9];
  const float* ln_b         = (const float*)d_in[10];
  const float* final_norm_w = (const float*)d_in[11];
  const float* head_W       = (const float*)d_in[12];
  const float* head_b       = (const float*)d_in[13];

  char* wsp = (char*)d_ws;
  size_t off = 0;
  auto alloc = [&](size_t bytes) -> void* {
    void* p = wsp + off;
    off += (bytes + 255) & ~(size_t)255;
    return p;
  };
  float*          x     = (float*)alloc((size_t)M_ * D_ * 4);
  float*          nx    = (float*)alloc((size_t)M_ * D_ * 4);
  unsigned short* nxb   = (unsigned short*)alloc((size_t)M_ * D_ * 2);
  float*          xg    = (float*)alloc((size_t)M_ * G_ * 4);
  unsigned short* hall  = (unsigned short*)alloc((size_t)M_ * H_ * 2);
  float*          outb  = (float*)alloc((size_t)M_ * D_ * 4);
  unsigned short* hcur  = (unsigned short*)alloc((size_t)B_ * H_ * 2);
  unsigned short* wihb  = (unsigned short*)alloc((size_t)G_ * D_ * 2);
  unsigned short* whhb  = (unsigned short*)alloc((size_t)G_ * H_ * 2);
  unsigned short* woutb = (unsigned short*)alloc((size_t)D_ * H_ * 2);
  unsigned short* headb = (unsigned short*)alloc((size_t)V_ * D_ * 2);
  unsigned short* lastb = (unsigned short*)alloc((size_t)B_ * D_ * 2);
  float*          biasc = (float*)alloc((size_t)G_ * 4);
  unsigned*       bar   = (unsigned*)alloc(256);

  hipMemsetAsync(bar, 0, 256, stream);
  k_embed<<<(M_ * D_) / 256, 256, 0, stream>>>(tokens, embedding, x);

  for (int l = 0; l < L_; ++l) {
    k_rmsnorm<<<M_, 256, 0, stream>>>(x, pre_norm_w + l * D_, nx, nxb);
    k_f2bf<<<(G_ * D_) / 256, 256, 0, stream>>>(W_ih + (size_t)l * G_ * D_, wihb, G_ * D_);
    k_f2bf<<<(G_ * H_) / 256, 256, 0, stream>>>(W_hh + (size_t)l * G_ * H_, whhb, G_ * H_);
    k_f2bf<<<(D_ * H_) / 256, 256, 0, stream>>>(W_out + (size_t)l * D_ * H_, woutb, D_ * H_);
    k_biascomb<<<G_ / 256, 256, 0, stream>>>(b_ih + l * G_, b_hh + l * G_, biasc);

    // xg[b,s,:] = nx @ W_ih^T + (b_ih + b_hh)
    k_gemm_bf16<<<dim3(M_ / 16, G_ / 128), 256, 0, stream>>>(nxb, D_, wihb, D_, biasc, xg, G_, D_);

    hipMemsetAsync(hcur, 0, (size_t)B_ * H_ * 2, stream);
    k_lstm<<<H_ / 16, 256, 0, stream>>>(xg, whhb, hcur, hall, bar, bar + 1);

    // out = h_all @ W_out^T + b_out
    k_gemm_bf16<<<dim3(M_ / 16, D_ / 128), 256, 0, stream>>>(hall, H_, woutb, H_,
                                                             b_out + l * D_, outb, D_, H_);
    k_add_layernorm<<<M_, 256, 0, stream>>>(outb, nx, ln_w + l * D_, ln_b + l * D_, x);
  }

  k_finalnorm<<<B_, 256, 0, stream>>>(x, final_norm_w, lastb);
  k_f2bf<<<(V_ * D_) / 256, 256, 0, stream>>>(head_W, headb, V_ * D_);
  // logits = last @ head_W^T + head_b
  k_gemm_bf16<<<dim3(B_ / 16, V_ / 128), 256, 0, stream>>>(lastb, D_, headb, D_,
                                                           head_b, (float*)d_out, V_, D_);
}
